// WSMultiQueryAttention_56495999811808
// MI455X (gfx1250) — compile-verified, matrix operand order checked by reading, not run on hardware
//
#include <hip/hip_runtime.h>

typedef __attribute__((ext_vector_type(16))) __bf16       v16bf;
typedef __attribute__((ext_vector_type(8)))  float        v8f;
typedef __attribute__((ext_vector_type(4)))  unsigned int v4u;
typedef __attribute__((ext_vector_type(8)))  int          v8i;
typedef __attribute__((ext_vector_type(4)))  int          v4i;

constexpr int D_MODEL = 2048;
constexpr int N_HEADS = 16;
constexpr int D_HEAD  = 128;
constexpr int BATCH   = 2;
constexpr int SEQ     = 2048;
constexpr int QKV_N   = D_MODEL + 2 * D_HEAD;   // 2304
constexpr int MROWS   = BATCH * SEQ;            // 4096
constexpr float SCALE = 0.08838834764831845f;   // 1/sqrt(128), folded into q

// LDS B-tile row pitch: 32 data elements (64B) + 8 pad elements (16B)
// matches TDM pad_interval=64B / pad_amount=16B and de-conflicts LDS banks.
constexpr int ROWPAD = 40;

#if defined(__has_builtin)
#  if __has_builtin(__builtin_amdgcn_tensor_load_to_lds) && \
      __has_builtin(__builtin_amdgcn_s_wait_tensorcnt)
#    define HAVE_TDM 1
#  endif
#endif
#ifndef HAVE_TDM
#  define HAVE_TDM 0
#endif

union bf16x16 { v4u u[2]; v16bf v; };

static __device__ __forceinline__ __bf16 f2bf(float f) {
    unsigned int u = __builtin_bit_cast(unsigned int, f);
    u += 0x7FFFu + ((u >> 16) & 1u);            // round-to-nearest-even
    unsigned short h = (unsigned short)(u >> 16);
    return __builtin_bit_cast(__bf16, h);
}

// Load a 16x32 bf16 fragment in the WMMA A/B VGPR layout:
// lanes 0-15 : row r = lane,    K = 0..7 / 16..23
// lanes 16-31: row r = lane-16, K = 8..15 / 24..31
static __device__ __forceinline__ v16bf load_frag(const __bf16* base, int stride) {
    int lane = threadIdx.x & 31;
    int r    = lane & 15;
    int kb   = (lane >> 4) << 3;
    const __bf16* p = base + (size_t)r * stride + kb;
    bf16x16 t;
    t.u[0] = *(const v4u*)(p);
    t.u[1] = *(const v4u*)(p + 16);
    return t.v;
}

static __device__ __forceinline__ v8f wmma_bf16(v16bf a, v16bf b, v8f c) {
    return __builtin_amdgcn_wmma_f32_16x16x32_bf16(false, a, false, b,
                                                   (short)0, c, false, false);
}

#if HAVE_TDM
// TDM: DMA a 2D bf16 tile (32 K-elems x 64 N-rows, row stride = D_MODEL elems)
// from global into LDS at lds_off, inserting 16B of LDS padding every 64B.
static __device__ __forceinline__ void tdm_stage_B(const __bf16* gsrc, unsigned lds_off) {
    unsigned long long ga = (unsigned long long)(size_t)gsrc;
    const unsigned TD = 1u << 30;               // huge tensor dims -> no OOB clip
    v4u g0 = { 1u,                              // count=1, is_restore=0
               lds_off,                         // lds_addr (bytes, WG-relative)
               (unsigned)ga,                    // global_addr[31:0]
               (unsigned)((ga >> 32) & 0x01FFFFFFu) | (2u << 30) };  // [56:32] | type=2
    v8i g1 = { (int)((1u << 16)                 // data_size = 2 bytes
                   | (1u << 20)                 // pad_enable
                   | (3u << 22)                 // pad_interval: 16 DWORDs = 64B
                   | (3u << 25)),               // pad_amount:    4 DWORDs = 16B
               (int)((TD & 0xFFFFu) << 16),                       // tensor_dim0[15:0]
               (int)(((TD >> 16) & 0xFFFFu) | ((TD & 0xFFFFu) << 16)), // td0 hi | td1 lo
               (int)(((TD >> 16) & 0xFFFFu) | (32u << 16)),       // td1 hi | tile_dim0=32
               64,                                                // tile_dim1=64, tile_dim2=0
               (int)(unsigned)D_MODEL,                            // tensor_dim0_stride lo
               0, 0 };
    v4i z4 = {};
#if __clang_major__ >= 23
    v8i z8 = {};
    __builtin_amdgcn_tensor_load_to_lds(g0, g1, z4, z4, z8, 0);
#else
    __builtin_amdgcn_tensor_load_to_lds(g0, g1, z4, z4, 0);
#endif
}
#else
// Cooperative fallback: 128 threads stage one 64(N) x 32(K) bf16 tile.
static __device__ __forceinline__ void coop_stage_B(__bf16* dst_lds, const __bf16* gsrc,
                                                    int tid) {
    int r  = tid >> 1;
    int hh = tid & 1;
    const __bf16* src = gsrc + (size_t)r * D_MODEL + hh * 16;
    v4u* d = (v4u*)(dst_lds + r * ROWPAD + hh * 16);
    d[0] = *(const v4u*)(src);
    d[1] = *(const v4u*)(src + 8);
}
#endif

// ---------------- fp32 -> bf16 bulk convert --------------------------------
__global__ __launch_bounds__(256) void conv_bf16_kernel(const float* __restrict__ src,
                                                        __bf16* __restrict__ dst,
                                                        int n) {
    int i = (blockIdx.x * 256 + threadIdx.x) * 4;
    if (i + 3 < n) {
        float4 f = *(const float4*)(src + i);
        dst[i + 0] = f2bf(f.x);
        dst[i + 1] = f2bf(f.y);
        dst[i + 2] = f2bf(f.z);
        dst[i + 3] = f2bf(f.w);
    }
}

// ---------------- fp32 [K x N] -> bf16 [N x K] transpose -------------------
__global__ __launch_bounds__(256) void transpose_conv_kernel(const float* __restrict__ W,
                                                             __bf16* __restrict__ WT,
                                                             int Krows, int Ncols) {
    __shared__ __bf16 tile[32][33];
    int ntiles = Ncols >> 5;
    int bx = blockIdx.x % ntiles;
    int by = blockIdx.x / ntiles;
    int tx = threadIdx.x & 31;
    int ty = threadIdx.x >> 5;
#pragma unroll
    for (int i = 0; i < 32; i += 8) {
        int k  = by * 32 + ty + i;
        int nn = bx * 32 + tx;
        tile[ty + i][tx] = f2bf(W[(size_t)k * Ncols + nn]);
    }
    __syncthreads();
#pragma unroll
    for (int i = 0; i < 32; i += 8) {
        int nn = bx * 32 + ty + i;
        int k  = by * 32 + tx;
        WT[(size_t)nn * Krows + k] = tile[tx][ty + i];
    }
}

// ---------------- GEMM core: block = 4 waves = 128x64 tile -----------------
// B tile DMA'd by the Tensor Data Mover into LDS (double buffered, issued one
// step ahead, synchronized with s_wait_tensorcnt); A register double-buffered;
// each wave computes 32x64 with 8 WMMAs per 32-K step. Last step peeled so the
// hot loop has no conditionals. MODE 0: QKV epilogue (bf16 split into q/k/vT,
// q pre-scaled by 1/sqrt(Dh)). MODE 1: fp32 + bias to output.
template<int NCOLS, int MODE>
static __device__ __forceinline__ void gemm_body(const __bf16* __restrict__ A,
                                                 const __bf16* __restrict__ BT,
                                                 const float* __restrict__ bias,
                                                 float* __restrict__ outf,
                                                 __bf16* __restrict__ qb,
                                                 __bf16* __restrict__ kbuf,
                                                 __bf16* __restrict__ vtb) {
    __shared__ __bf16 Bbuf[2][64 * ROWPAD];
    const int NT   = NCOLS / 64;
    const int br   = blockIdx.x / NT;
    const int nt   = blockIdx.x % NT;
    const int wave = threadIdx.x >> 5;
    const int lane = threadIdx.x & 31;
    const int n    = lane & 15;
    const int half = lane >> 4;
    const int kb   = half << 3;
    const int row0 = br * 128 + wave * 32;
    const int steps = D_MODEL / 32;              // 64

    const __bf16* Btile = BT + (size_t)(nt * 64) * D_MODEL;
    const __bf16* Arow0 = A + (size_t)row0 * D_MODEL;

    v8f acc0[4], acc1[4];
#pragma unroll
    for (int s = 0; s < 4; ++s) { acc0[s] = v8f{}; acc1[s] = v8f{}; }

    // stage step 0; preload A fragments for step 0
#if HAVE_TDM
    if (wave == 0) tdm_stage_B(Btile, (unsigned)(size_t)&Bbuf[0][0]);
#else
    coop_stage_B(&Bbuf[0][0], Btile, threadIdx.x);
#endif
    v16bf a0c = load_frag(Arow0, D_MODEL);
    v16bf a1c = load_frag(Arow0 + (size_t)16 * D_MODEL, D_MODEL);

#pragma unroll 1
    for (int j = 0; j < steps - 1; ++j) {
        const int cur = j & 1;
#if HAVE_TDM
        if (wave == 0) {
            tdm_stage_B(Btile + (j + 1) * 32, (unsigned)(size_t)&Bbuf[cur ^ 1][0]);
            __builtin_amdgcn_s_wait_tensorcnt(1);   // step j's DMA complete
        }
#else
        coop_stage_B(&Bbuf[cur ^ 1][0], Btile + (j + 1) * 32, threadIdx.x);
#endif
        __syncthreads();
        // register double-buffer A: issue step j+1 loads now, consume next iter
        v16bf a0n = load_frag(Arow0 + (j + 1) * 32, D_MODEL);
        v16bf a1n = load_frag(Arow0 + (size_t)16 * D_MODEL + (j + 1) * 32, D_MODEL);
        // all 4 B fragments into distinct registers, then 8 WMMAs unblocked
        v16bf bfr[4];
#pragma unroll
        for (int s = 0; s < 4; ++s) {
            const __bf16* p = &Bbuf[cur][(s * 16 + n) * ROWPAD + kb];
            bf16x16 t;
            t.u[0] = *(const v4u*)(p);
            t.u[1] = *(const v4u*)(p + 16);
            bfr[s] = t.v;
        }
#pragma unroll
        for (int s = 0; s < 4; ++s) acc0[s] = wmma_bf16(a0c, bfr[s], acc0[s]);
#pragma unroll
        for (int s = 0; s < 4; ++s) acc1[s] = wmma_bf16(a1c, bfr[s], acc1[s]);
        a0c = a0n;
        a1c = a1n;
        __syncthreads();
    }
    {   // peeled final step (buffer parity (steps-1)&1 == 1)
        const int cur = (steps - 1) & 1;
#if HAVE_TDM
        if (wave == 0) __builtin_amdgcn_s_wait_tensorcnt(0);
#endif
        __syncthreads();
        v16bf bfr[4];
#pragma unroll
        for (int s = 0; s < 4; ++s) {
            const __bf16* p = &Bbuf[cur][(s * 16 + n) * ROWPAD + kb];
            bf16x16 t;
            t.u[0] = *(const v4u*)(p);
            t.u[1] = *(const v4u*)(p + 16);
            bfr[s] = t.v;
        }
#pragma unroll
        for (int s = 0; s < 4; ++s) acc0[s] = wmma_bf16(a0c, bfr[s], acc0[s]);
#pragma unroll
        for (int s = 0; s < 4; ++s) acc1[s] = wmma_bf16(a1c, bfr[s], acc1[s]);
    }

#pragma unroll
    for (int s = 0; s < 4; ++s) {
        int col  = nt * 64 + s * 16 + n;
        float bv = bias[col];
#pragma unroll
        for (int m = 0; m < 2; ++m) {
            v8f accv = m ? acc1[s] : acc0[s];
#pragma unroll
            for (int v = 0; v < 8; ++v) {
                int row = row0 + m * 16 + half * 8 + v;
                float val = accv[v] + bv;
                if constexpr (MODE == 1) {
                    outf[(size_t)row * NCOLS + col] = val;
                } else {
                    if (col < D_MODEL) {
                        qb[(size_t)row * D_MODEL + col] = f2bf(val * SCALE);
                    } else {
                        int b = row / SEQ, sq = row % SEQ;
                        if (col < D_MODEL + D_HEAD) {
                            kbuf[((size_t)b * SEQ + sq) * D_HEAD + (col - D_MODEL)] = f2bf(val);
                        } else {
                            vtb[((size_t)b * D_HEAD + (col - D_MODEL - D_HEAD)) * SEQ + sq] = f2bf(val);
                        }
                    }
                }
            }
        }
    }
}

__global__ __launch_bounds__(128) void gemm_qkv_kernel(const __bf16* __restrict__ A,
                                                       const __bf16* __restrict__ BT,
                                                       const float* __restrict__ bias,
                                                       __bf16* __restrict__ qb,
                                                       __bf16* __restrict__ kbuf,
                                                       __bf16* __restrict__ vtb) {
    gemm_body<QKV_N, 0>(A, BT, bias, nullptr, qb, kbuf, vtb);
}

__global__ __launch_bounds__(128) void gemm_out_kernel(const __bf16* __restrict__ A,
                                                       const __bf16* __restrict__ BT,
                                                       const float* __restrict__ bias,
                                                       float* __restrict__ outf) {
    gemm_body<D_MODEL, 1>(A, BT, bias, outf, nullptr, nullptr, nullptr);
}

// ---------------- attention: one 32-key block ------------------------------
template<bool MASKED>
static __device__ __forceinline__ void attn_block(const __bf16* __restrict__ K,
                                                  const __bf16* __restrict__ VT,
                                                  __bf16* pl, const v16bf* qa,
                                                  v8f* acc, float* mrow, float* lrow,
                                                  int sk, int q0) {
    int lane = threadIdx.x & 31;
    int n    = lane & 15;
    int half = lane >> 4;

    v16bf kf[8];
#pragma unroll
    for (int ch = 0; ch < 4; ++ch) {
        kf[ch]     = load_frag(K + (size_t)sk * D_HEAD + ch * 32, D_HEAD);
        kf[4 + ch] = load_frag(K + (size_t)(sk + 16) * D_HEAD + ch * 32, D_HEAD);
    }
    v8f s0 = v8f{}, s1 = v8f{};
#pragma unroll
    for (int ch = 0; ch < 4; ++ch) {
        s0 = wmma_bf16(qa[ch], kf[ch],     s0);
        s1 = wmma_bf16(qa[ch], kf[4 + ch], s1);
    }
    float pr0[8], pr1[8];
#pragma unroll
    for (int v = 0; v < 8; ++v) {
        float x0 = s0[v];
        float x1 = s1[v];
        if constexpr (MASKED) {
            int row = q0 + half * 8 + v;
            if (sk + n      > row) x0 = -1e30f;
            if (sk + 16 + n > row) x1 = -1e30f;
        }
        float bm = fmaxf(x0, x1);
#pragma unroll
        for (int m = 1; m < 16; m <<= 1) bm = fmaxf(bm, __shfl_xor(bm, m, 32));
        float mnew  = fmaxf(mrow[v], bm);
        float alpha = __expf(mrow[v] - mnew);
        float p0 = __expf(x0 - mnew);
        float p1 = __expf(x1 - mnew);
        float bs = p0 + p1;
#pragma unroll
        for (int m = 1; m < 16; m <<= 1) bs += __shfl_xor(bs, m, 32);
        lrow[v] = lrow[v] * alpha + bs;
        mrow[v] = mnew;
        pr0[v]  = p0;
        pr1[v]  = p1;
#pragma unroll
        for (int tt = 0; tt < 8; ++tt) acc[tt][v] *= alpha;
    }
    __syncthreads();
#pragma unroll
    for (int v = 0; v < 8; ++v) {
        pl[(half * 8 + v) * 32 + n]      = f2bf(pr0[v]);
        pl[(half * 8 + v) * 32 + 16 + n] = f2bf(pr1[v]);
    }
    __syncthreads();
    v16bf pa = load_frag(pl, 32);
    v16bf vf[8];
#pragma unroll
    for (int tt = 0; tt < 8; ++tt)
        vf[tt] = load_frag(VT + (size_t)(tt * 16) * SEQ + sk, SEQ);
#pragma unroll
    for (int tt = 0; tt < 8; ++tt)
        acc[tt] = wmma_bf16(pa, vf[tt], acc[tt]);
}

// ---------------- Flash-style causal MQA attention -------------------------
__global__ __launch_bounds__(32) void attn_kernel(const __bf16* __restrict__ qbuf,
                                                  const __bf16* __restrict__ kbuf,
                                                  const __bf16* __restrict__ vtbuf,
                                                  __bf16* __restrict__ obuf) {
    const int QT = SEQ / 16;
    int t  = blockIdx.x;
    int qt = t % QT;  t /= QT;
    int h  = t % N_HEADS;
    int b  = t / N_HEADS;
    int q0 = qt * 16;

    int lane = threadIdx.x & 31;
    int n    = lane & 15;
    int half = lane >> 4;

    const __bf16* Q  = qbuf  + ((size_t)(b * SEQ + q0)) * D_MODEL + h * D_HEAD;
    const __bf16* K  = kbuf  + (size_t)b * SEQ * D_HEAD;
    const __bf16* VT = vtbuf + (size_t)b * D_HEAD * SEQ;

    v16bf qa[4];
#pragma unroll
    for (int ch = 0; ch < 4; ++ch) qa[ch] = load_frag(Q + ch * 32, D_MODEL);

    v8f acc[8];
#pragma unroll
    for (int i = 0; i < 8; ++i) acc[i] = v8f{};
    float mrow[8], lrow[8];
#pragma unroll
    for (int v = 0; v < 8; ++v) { mrow[v] = -1e30f; lrow[v] = 0.f; }

    __shared__ __bf16 pl[16 * 32];

    const int last = ((q0 + 15) >> 5) << 5;      // only block needing the mask
#pragma unroll 1
    for (int sk = 0; sk < last; sk += 32)
        attn_block<false>(K, VT, pl, qa, acc, mrow, lrow, sk, q0);
    attn_block<true>(K, VT, pl, qa, acc, mrow, lrow, last, q0);

#pragma unroll
    for (int v = 0; v < 8; ++v) {
        float inv = 1.0f / lrow[v];
        int row = b * SEQ + q0 + half * 8 + v;
#pragma unroll
        for (int tt = 0; tt < 8; ++tt) {
            obuf[(size_t)row * D_MODEL + h * D_HEAD + tt * 16 + n] = f2bf(acc[tt][v] * inv);
        }
    }
}

extern "C" void kernel_launch(void* const* d_in, const int* in_sizes, int n_in,
                              void* d_out, int out_size, void* d_ws, size_t ws_size,
                              hipStream_t stream) {
    const float* x     = (const float*)d_in[0];
    const float* W_qkv = (const float*)d_in[1];
    const float* b_qkv = (const float*)d_in[2];
    const float* W_out = (const float*)d_in[3];
    const float* b_out = (const float*)d_in[4];
    float* out = (float*)d_out;

    char* ws = (char*)d_ws;
    size_t off = 0;
    auto carve = [&](size_t bytes) {
        size_t r = off;
        off += (bytes + 255) & ~(size_t)255;
        return r;
    };
    __bf16* xbf   = (__bf16*)(ws + carve((size_t)MROWS * D_MODEL * 2));
    __bf16* WqkvT = (__bf16*)(ws + carve((size_t)QKV_N * D_MODEL * 2));
    __bf16* WoutT = (__bf16*)(ws + carve((size_t)D_MODEL * D_MODEL * 2));
    __bf16* qbuf  = (__bf16*)(ws + carve((size_t)MROWS * D_MODEL * 2));
    __bf16* kbuf  = (__bf16*)(ws + carve((size_t)BATCH * SEQ * D_HEAD * 2));
    __bf16* vtbuf = (__bf16*)(ws + carve((size_t)BATCH * D_HEAD * SEQ * 2));
    __bf16* obuf  = (__bf16*)(ws + carve((size_t)MROWS * D_MODEL * 2));
    (void)ws_size; (void)in_sizes; (void)n_in; (void)out_size;

    {   // x -> bf16
        int n = MROWS * D_MODEL;
        conv_bf16_kernel<<<n / (256 * 4), 256, 0, stream>>>(x, xbf, n);
    }
    transpose_conv_kernel<<<(D_MODEL / 32) * (QKV_N / 32), 256, 0, stream>>>(
        W_qkv, WqkvT, D_MODEL, QKV_N);
    transpose_conv_kernel<<<(D_MODEL / 32) * (D_MODEL / 32), 256, 0, stream>>>(
        W_out, WoutT, D_MODEL, D_MODEL);
    // QKV projection (TDM-staged B, 128x64 blocks)
    gemm_qkv_kernel<<<(MROWS / 128) * (QKV_N / 64), 128, 0, stream>>>(
        xbf, WqkvT, b_qkv, qbuf, kbuf, vtbuf);
    // causal MQA flash attention
    attn_kernel<<<BATCH * N_HEADS * (SEQ / 16), 32, 0, stream>>>(qbuf, kbuf, vtbuf, obuf);
    // output projection
    gemm_out_kernel<<<(MROWS / 128) * (D_MODEL / 64), 128, 0, stream>>>(
        obuf, WoutT, b_out, out);
}
